// GraphSim_12257836663008
// MI455X (gfx1250) — matrix operand order repaired and checked by reading, not compiled, not run on hardware
//
#include <hip/hip_runtime.h>
#include <hip/hip_bf16.h>
#include <math.h>

// ---------------- constants from the reference ----------------
#define GB   512        // graph pairs (B)
#define GM   64         // nodes per graph (M)
#define NN   (GB*GM)    // 32768 nodes per side
#define DIN  32
#define NE   (GB*512)   // edges per side = 262144

#define CDIV(a,b) (((a)+(b)-1)/(b))

typedef __attribute__((ext_vector_type(16))) __bf16 v16bf;
typedef __attribute__((ext_vector_type(8)))  __bf16 v8bf;
typedef __attribute__((ext_vector_type(8)))  float  v8f;
typedef unsigned short ushort_t;

__device__ __forceinline__ ushort_t f2bf(float f) {
    unsigned u = __float_as_uint(f);
    u += 0x7FFFu + ((u >> 16) & 1u);            // round-to-nearest-even
    return (ushort_t)(u >> 16);
}
__device__ __forceinline__ float bf2f(ushort_t u) {
    return __uint_as_float(((unsigned)u) << 16);
}
__device__ __forceinline__ __bf16 as_bf(ushort_t u) {
    __bf16 b; __builtin_memcpy(&b, &u, 2); return b;
}

// ---- WMMA fragment loaders (bf16, 16x16x32, wave32; ISA 7.12.2 layouts) ----
// A (16x32, row-major source): lane L holds row (L&15); halfs K = hi..hi+7 and
// K = 16+hi..16+hi+7 where hi = (L>>4)*8.
__device__ __forceinline__ v16bf load_a_frag(const ushort_t* A, int lda, int row, int k0, int lane) {
    int hi = (lane >> 4) * 8;
    const ushort_t* p = A + (size_t)row * lda + k0 + hi;
    v8bf lo = *(const v8bf*)(p);
    v8bf hi8 = *(const v8bf*)(p + 16);
    v16bf a;
#pragma unroll
    for (int i = 0; i < 8; ++i) { a[i] = lo[i]; a[i + 8] = hi8[i]; }
    return a;
}
// B (32x16) from Bt[N][K] (i.e. B^T row-major): lane L holds column (L&15),
// halfs K = hi..hi+7 and K = 16+hi..16+hi+7 where hi = (L>>4)*8.
__device__ __forceinline__ v16bf load_b_frag(const ushort_t* Bt, int ldb, int col, int k0, int lane) {
    int hi = (lane >> 4) * 8;
    const ushort_t* p = Bt + (size_t)col * ldb + k0 + hi;
    v8bf lo = *(const v8bf*)(p);
    v8bf hi8 = *(const v8bf*)(p + 16);
    v16bf b;
#pragma unroll
    for (int i = 0; i < 8; ++i) { b[i] = lo[i]; b[i + 8] = hi8[i]; }
    return b;
}

// ---------------- elementwise / prep kernels ----------------
__global__ void gs_zero32(unsigned int* p, long long n) {
    long long i = (long long)blockIdx.x * blockDim.x + threadIdx.x;
    if (i < n) p[i] = 0u;
}
// x [rows][srcw] f32 -> dst [rows][dstw] bf16 (cols < srcw)
__global__ void gs_cvt_x(ushort_t* dst, const float* src, int rows, int srcw, int dstw) {
    int i = blockIdx.x * blockDim.x + threadIdx.x;
    if (i >= rows * srcw) return;
    int r = i / srcw, c = i % srcw;
    dst[(size_t)r * dstw + c] = f2bf(src[i]);
}
// W [K][N] f32 -> Wt [N][K] bf16
__global__ void gs_prep_wT(ushort_t* dst, const float* src, int K, int N) {
    long long i = (long long)blockIdx.x * blockDim.x + threadIdx.x;
    if (i >= (long long)K * N) return;
    int n = (int)(i / K), k = (int)(i % K);
    dst[i] = f2bf(src[(size_t)k * N + n]);
}
// conv weights (NI, OC, Cin, 5, 5) f32 -> Bt [NI][16][Kpad] bf16, k = pos*Cin+ic
__global__ void gs_prep_cw(ushort_t* dst, const float* src, int NI, int OC, int Cin, int Kpad, int Kreal) {
    int i = blockIdx.x * blockDim.x + threadIdx.x;
    int tot = NI * 16 * Kpad;
    if (i >= tot) return;
    int k = i % Kpad, oc = (i / Kpad) % 16, im = i / (Kpad * 16);
    float v = 0.f;
    if (oc < OC && k < Kreal) {
        int ic = k % Cin, pos = k / Cin;
        v = src[(((size_t)(im * OC + oc) * Cin + ic) * 25) + pos];
    }
    dst[i] = f2bf(v);
}

// ---------------- GCN helpers ----------------
__global__ void gs_deg(float* deg, const int* dst, int nE) {
    int e = blockIdx.x * blockDim.x + threadIdx.x;
    if (e < nE) atomicAdd(&deg[dst[e]], 1.0f);
}
__global__ void gs_dinv(float* deg, int n) {   // deg -> rsqrt(deg + 1 self loop)
    int i = blockIdx.x * blockDim.x + threadIdx.x;
    if (i < n) deg[i] = rsqrtf(deg[i] + 1.0f);
}
// OUT[n][c] = bias[c] + H[n][c] * dinv[n]^2   (self-loop term)
__global__ void gs_gcn_init(float* OUT, const float* H, const float* dinv, const float* bias, int od, int n) {
    int i = blockIdx.x * blockDim.x + threadIdx.x;
    if (i >= n) return;
    int node = i / od, c = i % od;
    float dv = dinv[node];
    OUT[i] = bias[c] + H[i] * dv * dv;
}
// OUT[d][c] += H[s][c] * dinv[s]*dinv[d]   (one thread per (edge, channel))
__global__ void gs_gcn_edge(float* OUT, const float* H, const int* src, const int* dst,
                            const float* dinv, int od, int nEC) {
    int i = blockIdx.x * blockDim.x + threadIdx.x;
    if (i >= nEC) return;
    int e = i / od, c = i % od;
    int s = src[e], d = dst[e];
    float nm = dinv[s] * dinv[d];
    atomicAdd(&OUT[(size_t)d * od + c], H[(size_t)s * od + c] * nm);
}
// feats[n][c<fd] = bf16(OUT[n][c<od] else 0);  Xnext[n][c<od] = bf16(relu(OUT))
__global__ void gs_gcn_fin(ushort_t* feats, ushort_t* Xnext, const float* OUT, int od, int fd, int n) {
    int i = blockIdx.x * blockDim.x + threadIdx.x;
    if (i >= n) return;
    int node = i / fd, c = i % fd;
    float v = (c < od) ? OUT[(size_t)node * od + c] : 0.f;
    feats[i] = f2bf(v);
    if (Xnext && c < od) Xnext[(size_t)node * 64 + c] = f2bf(fmaxf(v, 0.f));
}

// ---------------- generic WMMA GEMM ----------------
// C[M][N](ldc) = A[M][K](lda, bf16) * Bt[N][K](ldb, bf16)^T (+bias)
// mode 0: f32 out; 1: f32 relu; 2: bf16 relu
__global__ void gs_gemm(const ushort_t* A, int lda, const ushort_t* Bt, int ldb,
                        const float* bias, void* Cout, int ldc, int K, int mode) {
    int lane = threadIdx.x;
    int n0 = blockIdx.x * 16, m0 = blockIdx.y * 16;
    v8f acc = {};
    for (int k0 = 0; k0 < K; k0 += 32) {
        v16bf a = load_a_frag(A, lda, m0 + (lane & 15), k0, lane);
        v16bf b = load_b_frag(Bt, ldb, n0 + (lane & 15), k0, lane);
        acc = __builtin_amdgcn_wmma_f32_16x16x32_bf16(false, a, false, b, (short)0, acc, false, false);
    }
    int col = n0 + (lane & 15);
    int rbase = m0 + ((lane >> 4) << 3);
    float bv = bias ? bias[col] : 0.f;
#pragma unroll
    for (int r = 0; r < 8; ++r) {
        float v = acc[r] + bv;
        if (mode >= 1) v = fmaxf(v, 0.f);
        size_t idx = (size_t)(rbase + r) * ldc + col;
        if (mode == 2) ((ushort_t*)Cout)[idx] = f2bf(v);
        else           ((float*)Cout)[idx] = v;
    }
}

// ---------------- batched similarity GEMM: S_b = Q_b * C_b^T ----------------
// FQ/FC: [GB][GM][fd] bf16; Spad: [GB][68][68] bf16 (already offset per sim idx)
__global__ void gs_sim(const ushort_t* FQ, const ushort_t* FC, ushort_t* Spad, int fd) {
    int lane = threadIdx.x;
    int t = blockIdx.x;              // 0..15
    int m0 = (t >> 2) * 16, n0 = (t & 3) * 16;
    int b = blockIdx.y;
    const ushort_t* A  = FQ + (size_t)b * GM * fd;
    const ushort_t* Bt = FC + (size_t)b * GM * fd;
    v8f acc = {};
    for (int k0 = 0; k0 < fd; k0 += 32) {
        v16bf a  = load_a_frag(A, fd, m0 + (lane & 15), k0, lane);
        v16bf bf = load_b_frag(Bt, fd, n0 + (lane & 15), k0, lane);
        acc = __builtin_amdgcn_wmma_f32_16x16x32_bf16(false, a, false, bf, (short)0, acc, false, false);
    }
    int col = n0 + (lane & 15);
    int rbase = m0 + ((lane >> 4) << 3);
    ushort_t* o = Spad + (size_t)b * 68 * 68;
#pragma unroll
    for (int r = 0; r < 8; ++r)
        o[(size_t)(rbase + r + 2) * 68 + (col + 2)] = f2bf(acc[r]);
}

// ------------- conv1: 5x5, Cin=1 -> OC=8, 64x64, LDS-staged halo -------------
// In: [GB][68][68] bf16 padded; Bt: [16][32]; block = 4 waves = one output row.
__global__ void gs_conv1(const ushort_t* In, const ushort_t* Bt, const float* bias,
                         ushort_t* Out, int /*unused*/) {
    __shared__ ushort_t lds[5 * 68];
    int tid = threadIdx.x, lane = tid & 31, wv = tid >> 5;
    int b = blockIdx.y, y = blockIdx.x;                 // output row
    const ushort_t* inb = In + (size_t)b * 68 * 68;
    // stage rows y..y+4 (padded coords) cooperatively, coalesced in cols
    for (int s = tid; s < 5 * 68; s += 128) {
        int row = s / 68, col = s % 68;
        lds[s] = inb[(size_t)(y + row) * 68 + col];
    }
    __syncthreads();
    int x = (wv << 4) + (lane & 15);                    // output col for A row
    int hi = (lane >> 4) * 8;
    v16bf a;
#pragma unroll
    for (int j = 0; j < 16; ++j) {
        int kk = (j < 8) ? (hi + j) : (16 + hi + (j - 8));
        ushort_t uv = 0;
        if (kk < 25) {
            int ky = kk / 5, kx = kk % 5;
            uv = lds[ky * 68 + x + kx];
        }
        a[j] = as_bf(uv);
    }
    v16bf bf = load_b_frag(Bt, 32, (lane & 15), 0, lane);
    v8f acc = {};
    acc = __builtin_amdgcn_wmma_f32_16x16x32_bf16(false, a, false, bf, (short)0, acc, false, false);
    int oc = lane & 15;
    if (oc >= 8) return;
    float bv = bias[oc];
    int rbase = (wv << 4) + ((lane >> 4) << 3);
#pragma unroll
    for (int r = 0; r < 8; ++r) {
        float v = fmaxf(acc[r] + bv, 0.f);
        Out[((size_t)(b * 8 + oc) * 64 + y) * 64 + (rbase + r)] = f2bf(v);
    }
}

// ------ conv2: 5x5, Cin=8 -> OC=16, 32x32, LDS halo in [row][col][ic] ------
// In: [GB][8][36][36] bf16 padded; Bt: [16][224] (k = pos*8+ic);
// block = 4 waves covering 2 output rows; fragment halves are ds_load_b128s.
__global__ void gs_conv2(const ushort_t* In, const ushort_t* Bt, const float* bias,
                         ushort_t* Out, int /*unused*/) {
    __shared__ ushort_t lds[6 * 36 * 8];                // [row][col][ic]
    int tid = threadIdx.x, lane = tid & 31, wv = tid >> 5;
    int b = blockIdx.y, y0 = blockIdx.x * 2;
    const ushort_t* inb = In + (size_t)b * 8 * 36 * 36;
    // stage rows y0..y0+5, transposing [ic][row][col] -> [row][col][ic]
    for (int s = tid; s < 6 * 36 * 8; s += 128) {
        int col = s % 36, row = (s / 36) % 6, ic = s / 216;
        lds[(row * 36 + col) * 8 + ic] = inb[((size_t)ic * 36 + (y0 + row)) * 36 + col];
    }
    __syncthreads();
    int dy = wv >> 1;                                   // 0 or 1 (output row within block)
    int x = ((wv & 1) << 4) + (lane & 15);              // output col for A row
    int hi = (lane >> 4) * 8;
    v8f acc = {};
    for (int k0 = 0; k0 < 224; k0 += 32) {
        v16bf a;
        int posA = (k0 + hi) >> 3;                      // 8 contiguous ic per pos
        if (posA < 25) {
            int ky = posA / 5, kx = posA % 5;
            v8bf v = *(const v8bf*)&lds[((dy + ky) * 36 + x + kx) * 8];
#pragma unroll
            for (int i = 0; i < 8; ++i) a[i] = v[i];
        } else {
#pragma unroll
            for (int i = 0; i < 8; ++i) a[i] = as_bf((ushort_t)0);
        }
        int posB = (k0 + 16 + hi) >> 3;
        if (posB < 25) {
            int ky = posB / 5, kx = posB % 5;
            v8bf v = *(const v8bf*)&lds[((dy + ky) * 36 + x + kx) * 8];
#pragma unroll
            for (int i = 0; i < 8; ++i) a[i + 8] = v[i];
        } else {
#pragma unroll
            for (int i = 0; i < 8; ++i) a[i + 8] = as_bf((ushort_t)0);
        }
        v16bf bf = load_b_frag(Bt, 224, (lane & 15), k0, lane);
        acc = __builtin_amdgcn_wmma_f32_16x16x32_bf16(false, a, false, bf, (short)0, acc, false, false);
    }
    int oc = lane & 15;
    float bv = bias[oc];
    int y = y0 + dy;
    int rbase = ((wv & 1) << 4) + ((lane >> 4) << 3);
#pragma unroll
    for (int r = 0; r < 8; ++r) {
        float v = fmaxf(acc[r] + bv, 0.f);
        Out[((size_t)(b * 16 + oc) * 32 + y) * 32 + (rbase + r)] = f2bf(v);
    }
}

// ---------------- 2x2 max pools (inputs already ReLU'd) ----------------
// In [GB][C][2H][2W] -> padded out [GB][C][H+4][W+4] at (+2,+2)
__global__ void gs_pool_pad(ushort_t* Outp, const ushort_t* In, int C, int H, int W) {
    int i = blockIdx.x * blockDim.x + threadIdx.x;
    if (i >= GB * C * H * W) return;
    int x = i % W, y = (i / W) % H, c = (i / (W * H)) % C, b = i / (W * H * C);
    const ushort_t* p = In + (((size_t)(b * C + c) * (2 * H) + 2 * y) * (2 * W) + 2 * x);
    float m = fmaxf(fmaxf(bf2f(p[0]), bf2f(p[1])),
                    fmaxf(bf2f(p[2 * W]), bf2f(p[2 * W + 1])));
    Outp[((size_t)(b * C + c) * (H + 4) + (y + 2)) * (W + 4) + (x + 2)] = f2bf(m);
}
// In [GB][16][32][32] -> hbuf[b][((simIdx*16+c)*16+y)*16+x]
__global__ void gs_pool_hbuf(ushort_t* hbuf, const ushort_t* In, int simIdx) {
    int i = blockIdx.x * blockDim.x + threadIdx.x;
    if (i >= GB * 16 * 16 * 16) return;
    int x = i % 16, y = (i / 16) % 16, c = (i / 256) % 16, b = i / 4096;
    const ushort_t* p = In + (((size_t)(b * 16 + c) * 32 + 2 * y) * 32 + 2 * x);
    float m = fmaxf(fmaxf(bf2f(p[0]), bf2f(p[1])), fmaxf(bf2f(p[32]), bf2f(p[33])));
    hbuf[(size_t)b * 12288 + ((size_t)(simIdx * 16 + c) * 16 + y) * 16 + x] = f2bf(m);
}

// ---------------- final score ----------------
__global__ void gs_score(const float* h2, const float* Wsc, const float* bsc, float* out) {
    int b = blockIdx.x * blockDim.x + threadIdx.x;
    if (b >= GB) return;
    float z = bsc[0];
    for (int c = 0; c < 64; ++c) z += h2[(size_t)b * 64 + c] * Wsc[c];
    out[b] = 1.f / (1.f + expf(-z));
}

// =======================================================================
extern "C" void kernel_launch(void* const* d_in, const int* in_sizes, int n_in,
                              void* d_out, int out_size, void* d_ws, size_t ws_size,
                              hipStream_t stream) {
    const float* x_q = (const float*)d_in[0];
    const float* x_c = (const float*)d_in[1];
    const int* src_q = (const int*)d_in[2];
    const int* dst_q = (const int*)d_in[3];
    const int* src_c = (const int*)d_in[4];
    const int* dst_c = (const int*)d_in[5];
    const float* Wg[3] = {(const float*)d_in[6], (const float*)d_in[8], (const float*)d_in[10]};
    const float* bg[3] = {(const float*)d_in[7], (const float*)d_in[9], (const float*)d_in[11]};
    const float* cw0 = (const float*)d_in[12];
    const float* cb0 = (const float*)d_in[13];
    const float* cw1 = (const float*)d_in[14];
    const float* cb1 = (const float*)d_in[15];
    const float* Wl0 = (const float*)d_in[16];
    const float* bl0 = (const float*)d_in[17];
    const float* Wl1 = (const float*)d_in[18];
    const float* bl1 = (const float*)d_in[19];
    const float* Wsc = (const float*)d_in[20];
    const float* bsc = (const float*)d_in[21];
    float* out = (float*)d_out;

    // ---- workspace bump allocator ----
    char* base = (char*)d_ws;
    size_t off = 0;
    auto alloc = [&](size_t bytes) -> void* {
        void* p = base + off;
        off = (off + bytes + 255) & ~(size_t)255;
        return p;
    };
    float*    dinv   = (float*)alloc(2 * NN * 4);            // [q|c]
    ushort_t* Xq     = (ushort_t*)alloc((size_t)NN * 64 * 2);
    ushort_t* Xc     = (ushort_t*)alloc((size_t)NN * 64 * 2);
    float*    Hbuf   = (float*)alloc((size_t)NN * 64 * 4);
    float*    OUTb   = (float*)alloc((size_t)NN * 64 * 4);
    int fdL[3] = {64, 32, 32};                                // padded feat strides
    int odL[3] = {64, 32, 16};
    int idL[3] = {DIN, 64, 32};
    ushort_t* fq[3]; ushort_t* fc[3];
    for (int l = 0; l < 3; ++l) { fq[l] = (ushort_t*)alloc((size_t)NN * fdL[l] * 2);
                                  fc[l] = (ushort_t*)alloc((size_t)NN * fdL[l] * 2); }
    ushort_t* WgT[3];
    for (int l = 0; l < 3; ++l) WgT[l] = (ushort_t*)alloc((size_t)odL[l] * idL[l] * 2);
    ushort_t* cw0t = (ushort_t*)alloc((size_t)3 * 16 * 32 * 2);
    ushort_t* cw1t = (ushort_t*)alloc((size_t)3 * 16 * 224 * 2);
    ushort_t* Wl0t = (ushort_t*)alloc((size_t)256 * 12288 * 2);
    ushort_t* Wl1t = (ushort_t*)alloc((size_t)64 * 256 * 2);
    ushort_t* Spad  = (ushort_t*)alloc((size_t)3 * GB * 68 * 68 * 2);
    ushort_t* pool1 = (ushort_t*)alloc((size_t)GB * 8 * 36 * 36 * 2);
    ushort_t* c1out = (ushort_t*)alloc((size_t)GB * 8 * 64 * 64 * 2);
    ushort_t* c2out = (ushort_t*)alloc((size_t)GB * 16 * 32 * 32 * 2);
    ushort_t* hbuf  = (ushort_t*)alloc((size_t)GB * 12288 * 2);
    ushort_t* h1    = (ushort_t*)alloc((size_t)GB * 256 * 2);
    float*    h2    = (float*)alloc((size_t)GB * 64 * 4);

    const int T = 256;
    // ---- degrees -> dinv (self loop included) ----
    {
        long long w = (long long)2 * NN;   // f32 words
        gs_zero32<<<CDIV(w, T), T, 0, stream>>>((unsigned int*)dinv, w);
        gs_deg<<<CDIV(NE, T), T, 0, stream>>>(dinv, dst_q, NE);
        gs_deg<<<CDIV(NE, T), T, 0, stream>>>(dinv + NN, dst_c, NE);
        gs_dinv<<<CDIV(2 * NN, T), T, 0, stream>>>(dinv, 2 * NN);
    }
    // ---- input + weight conversion ----
    gs_cvt_x<<<CDIV(NN * DIN, T), T, 0, stream>>>(Xq, x_q, NN, DIN, 64);
    gs_cvt_x<<<CDIV(NN * DIN, T), T, 0, stream>>>(Xc, x_c, NN, DIN, 64);
    for (int l = 0; l < 3; ++l)
        gs_prep_wT<<<CDIV(idL[l] * odL[l], T), T, 0, stream>>>(WgT[l], Wg[l], idL[l], odL[l]);
    gs_prep_cw<<<CDIV(3 * 16 * 32, T), T, 0, stream>>>(cw0t, cw0, 3, 8, 1, 32, 25);
    gs_prep_cw<<<CDIV(3 * 16 * 224, T), T, 0, stream>>>(cw1t, cw1, 3, 16, 8, 224, 200);
    gs_prep_wT<<<CDIV((long long)12288 * 256, T), T, 0, stream>>>(Wl0t, Wl0, 12288, 256);
    gs_prep_wT<<<CDIV(256 * 64, T), T, 0, stream>>>(Wl1t, Wl1, 256, 64);
    // zero padded sim + pool1 buffers (borders must be 0)
    {
        long long w = ((long long)3 * GB * 68 * 68 * 2) / 4;
        gs_zero32<<<CDIV(w, T), T, 0, stream>>>((unsigned int*)Spad, w);
        long long w2 = ((long long)GB * 8 * 36 * 36 * 2) / 4;
        gs_zero32<<<CDIV(w2, T), T, 0, stream>>>((unsigned int*)pool1, w2);
    }

    // ---- GCN: both sides, 3 layers ----
    for (int side = 0; side < 2; ++side) {
        ushort_t* X = side ? Xc : Xq;
        ushort_t** fe = side ? fc : fq;
        const int* srcI = side ? src_c : src_q;
        const int* dstI = side ? dst_c : dst_q;
        const float* dv = dinv + (size_t)side * NN;
        for (int l = 0; l < 3; ++l) {
            int id = idL[l], od = odL[l], fd = fdL[l];
            dim3 g(od / 16, NN / 16);
            gs_gemm<<<g, 32, 0, stream>>>(X, 64, WgT[l], id, nullptr, Hbuf, od, id, 0);
            gs_gcn_init<<<CDIV(NN * od, T), T, 0, stream>>>(OUTb, Hbuf, dv, bg[l], od, NN * od);
            gs_gcn_edge<<<CDIV((long long)NE * od, T), T, 0, stream>>>(OUTb, Hbuf, srcI, dstI, dv, od, NE * od);
            gs_gcn_fin<<<CDIV(NN * fd, T), T, 0, stream>>>(fe[l], (l < 2) ? X : nullptr, OUTb, od, fd, NN * fd);
        }
    }

    // ---- similarity matrices -> padded bf16 [3][GB][68][68] ----
    for (int i = 0; i < 3; ++i) {
        dim3 g(16, GB);
        gs_sim<<<g, 32, 0, stream>>>(fq[i], fc[i], Spad + (size_t)i * GB * 68 * 68, fdL[i]);
    }

    // ---- per-sim CNN: conv1 -> pool -> conv2 -> pool ----
    for (int i = 0; i < 3; ++i) {
        dim3 g1(64, GB);   // one block per output row
        gs_conv1<<<g1, 128, 0, stream>>>(Spad + (size_t)i * GB * 68 * 68,
                                         cw0t + (size_t)i * 16 * 32,
                                         cb0 + (size_t)i * 8, c1out, 0);
        gs_pool_pad<<<CDIV(GB * 8 * 32 * 32, T), T, 0, stream>>>(pool1, c1out, 8, 32, 32);
        dim3 g2(16, GB);   // one block per 2 output rows
        gs_conv2<<<g2, 128, 0, stream>>>(pool1, cw1t + (size_t)i * 16 * 224,
                                         cb1 + (size_t)i * 16, c2out, 0);
        gs_pool_hbuf<<<CDIV(GB * 16 * 16 * 16, T), T, 0, stream>>>(hbuf, c2out, i);
    }

    // ---- MLP head ----
    {
        dim3 g0(256 / 16, GB / 16);
        gs_gemm<<<g0, 32, 0, stream>>>(hbuf, 12288, Wl0t, 12288, bl0, h1, 256, 12288, 2);
        dim3 g1(64 / 16, GB / 16);
        gs_gemm<<<g1, 32, 0, stream>>>(h1, 256, Wl1t, 256, bl1, h2, 64, 256, 1);
        gs_score<<<CDIV(GB, T), T, 0, stream>>>(h2, Wsc, bsc, out);
    }
}
